// AlignTTS_7481833029795
// MI455X (gfx1250) — matrix operand order repaired and checked by reading, not compiled, not run on hardware
//
#include <hip/hip_runtime.h>
#include <hip/hip_bf16.h>

typedef __attribute__((ext_vector_type(16))) _Float16 v16h;
typedef __attribute__((ext_vector_type(8)))  float    v8f;

#define BB 4
#define HH 256
#define CC 80
#define TX 256
#define TY 1024
#define KTOT 160              // 2*C
#define NFRAG 5               // KTOT / 32
#define NEGV (-1e9f)

// ---------------- workspace layout (bytes) ----------------
// Apack: B * (TX/16) tiles * 5 frags * 512 halves  = 163840 halves = 327680 B
// Bpack: B * (TY/16) tiles * 5 frags * 512 halves  = 655360 halves = 1310720 B
// constx: B*TX floats = 4096 B
// Xmap:   B*TY ints   = 16384 B
#define WS_A_OFF     0
#define WS_B_OFF     327680
#define WS_CONST_OFF 1638400
#define WS_XMAP_OFF  1642496

// 16-bit A-matrix 16x32 fragment, contiguous K order (matches B):
// lanes 0-15:  j=0..15 -> K=j      (K 0-15)
// lanes 16-31: j=0..15 -> K=j+16   (K 16-31)
__device__ __forceinline__ int apack_idx(int K, int m16) {
    int f = K >> 5, k32 = K & 31;
    int hi = k32 >> 4, j = k32 & 15;
    return f * 512 + (hi * 16 + m16) * 16 + j;
}

// 16-bit B-matrix 32x16: lanes 0-15 hold K=0..15 (j=K), lanes 16-31 hold K=16..31
__device__ __forceinline__ int bpack_idx(int K, int n16) {
    int f = K >> 5, k32 = K & 31;
    int hi = k32 >> 4, j = k32 & 15;
    return f * 512 + (hi * 16 + n16) * 16 + j;
}

// ---- Kernel 1a: pack A = [w(c), mu(c)*w(c)] (f16, fragment-swizzled) + constx
__global__ void k_pack_a(const float* __restrict__ mu,
                         const float* __restrict__ ls,
                         _Float16* __restrict__ Apack,
                         float* __restrict__ constx) {
    int tid = blockIdx.x * blockDim.x + threadIdx.x;      // B*TX
    if (tid >= BB * TX) return;
    int b = tid >> 8, x = tid & 255;
    int xt = x >> 4, m16 = x & 15;
    _Float16* Ab = Apack + (size_t)((b * (TX / 16) + xt) * NFRAG) * 512;
    float s2 = 0.f, sumls = 0.f;
    for (int c = 0; c < CC; ++c) {
        float m = mu[((size_t)(b * CC + c)) * TX + x];
        float l = ls[((size_t)(b * CC + c)) * TX + x];
        float w = __expf(-2.0f * l);
        Ab[apack_idx(c,       m16)] = (_Float16)w;
        Ab[apack_idx(CC + c,  m16)] = (_Float16)(m * w);
        s2 += m * m * w;
        sumls += l;
    }
    constx[tid] = (-0.5f / (float)CC) * (s2 + sumls);
}

// ---- Kernel 1b: pack B = [y(c)^2, -2*y(c)] (f16, fragment-swizzled)
__global__ void k_pack_b(const float* __restrict__ yv,
                         _Float16* __restrict__ Bpack) {
    int tid = blockIdx.x * blockDim.x + threadIdx.x;      // B*TY
    if (tid >= BB * TY) return;
    int b = tid >> 10, y = tid & 1023;
    int yt = y >> 4, n16 = y & 15;
    _Float16* Bb = Bpack + (size_t)((b * (TY / 16) + yt) * NFRAG) * 512;
    for (int c = 0; c < CC; ++c) {
        float v = yv[((size_t)(b * CC + c)) * TY + y];
        Bb[bpack_idx(c,      n16)] = (_Float16)(v * v);
        Bb[bpack_idx(CC + c, n16)] = (_Float16)(-2.0f * v);
    }
}

// ---- Kernel 2: logp via WMMA. One wave per 16x16 (x,y) tile, K=160 in 5 steps.
__global__ __launch_bounds__(256)
void k_logp_wmma(const _Float16* __restrict__ Apack,
                 const _Float16* __restrict__ Bpack,
                 const float* __restrict__ constx,
                 float* __restrict__ logp) {
    int w    = blockIdx.x * 8 + (threadIdx.x >> 5);       // global wave id
    int lane = threadIdx.x & 31;
    int b  = w >> 10;                                     // (TX/16)*(TY/16)=1024 tiles per b
    int xt = (w >> 6) & 15;
    int yt = w & 63;
    const _Float16* Ab = Apack + (size_t)((b * (TX / 16) + xt) * NFRAG) * 512;
    const _Float16* Bb = Bpack + (size_t)((b * (TY / 16) + yt) * NFRAG) * 512;
    v8f acc = {};
#pragma unroll
    for (int f = 0; f < NFRAG; ++f) {
        v16h a  = *reinterpret_cast<const v16h*>(Ab + f * 512 + lane * 16);
        v16h bm = *reinterpret_cast<const v16h*>(Bb + f * 512 + lane * 16);
        acc = __builtin_amdgcn_wmma_f32_16x16x32_f16(
            /*neg_a=*/false, a, /*neg_b=*/false, bm,
            /*c_mod=*/(short)0, acc, /*reuse_a=*/false, /*reuse_b=*/false);
    }
    const float scale = -0.5f / (float)CC;
    int x0 = xt * 16, y0 = yt * 16;
    int N  = lane & 15;
    int Mb = (lane >> 4) * 8;                             // C/D layout: lanes>=16 -> M+8
#pragma unroll
    for (int r = 0; r < 8; ++r) {
        int x = x0 + Mb + r;
        logp[((size_t)(b * TX + x)) * TY + y0 + N] = acc[r] * scale + constx[b * TX + x];
    }
}

// ---- Kernel 3: monotonic alignment DP (one block per batch, 256 threads = Tx)
__global__ __launch_bounds__(256)
void k_mas(const float* __restrict__ logp,
           const int* __restrict__ xlens,
           const int* __restrict__ ylens,
           float* __restrict__ attn,
           float* __restrict__ dr,
           int* __restrict__ Xmap) {
    __shared__ float sv[TX];
    __shared__ float dur[TX];
    __shared__ unsigned diagbits[TY * (TX / 32)];         // 32 KB
    int b = blockIdx.x;
    int x = threadIdx.x;
    int xlen = xlens[b], ylen = ylens[b];
    const float* lp = logp + ((size_t)(b * TX + x)) * TY;
    bool xact = x < xlen;
    float v = (x == 0) ? 0.0f : NEGV;

    for (int y = 0; y < TY; ++y) {
        float col = (xact && y < ylen) ? lp[y] : NEGV;
        sv[x] = v;
        __syncthreads();
        float vs = (x > 0) ? sv[x - 1] : NEGV;
        bool diag = vs > v;
        unsigned long long m = __ballot(diag);            // wave32: low 32 bits valid
        if ((x & 31) == 0) diagbits[y * (TX / 32) + (x >> 5)] = (unsigned)m;
        v = col + fmaxf(v, vs);
        __syncthreads();
    }

    // zero this batch's attn slice and duration accumulators
    dur[x] = 0.0f;
    float* ab = attn + ((size_t)(b * TX + x)) * TY;
    for (int y = 0; y < TY; ++y) ab[y] = 0.0f;
    __syncthreads();

    if (x == 0) {
        int idx = xlen - 1;
        for (int y = TY - 1; y >= 0; --y) {
            bool active = y < ylen;
            if (active) {
                attn[((size_t)(b * TX + idx)) * TY + y] = 1.0f;
                dur[idx] += 1.0f;
                Xmap[b * TY + y] = idx;
            } else {
                Xmap[b * TY + y] = -1;
            }
            unsigned dbit = (diagbits[y * (TX / 32) + (idx >> 5)] >> (idx & 31)) & 1u;
            if (active && idx > 0 && dbit) --idx;
        }
    }
    __syncthreads();
    dr[b * TX + x] = dur[x];
}

// ---- Kernel 4: o_en_ex[b,h,y] = en[b,h,Xmap[b,y]] (path == one-hot attn gather)
__global__ void k_expand(const float* __restrict__ en,
                         const int* __restrict__ Xmap,
                         float* __restrict__ o) {
    int tid = blockIdx.x * blockDim.x + threadIdx.x;      // B*H*TY
    if (tid >= BB * HH * TY) return;
    int y  = tid & (TY - 1);
    int bh = tid >> 10;                                   // b*H + h
    int b  = bh >> 8;                                     // H = 256
    int xi = Xmap[b * TY + y];
    o[tid] = (xi >= 0) ? en[(size_t)bh * TX + xi] : 0.0f;
}

extern "C" void kernel_launch(void* const* d_in, const int* in_sizes, int n_in,
                              void* d_out, int out_size, void* d_ws, size_t ws_size,
                              hipStream_t stream) {
    const float* en  = (const float*)d_in[0];
    const float* mu  = (const float*)d_in[1];
    const float* ls  = (const float*)d_in[2];
    const float* yv  = (const float*)d_in[3];
    const int*  xlen = (const int*)d_in[4];
    const int*  ylen = (const int*)d_in[5];

    float* out   = (float*)d_out;
    float* o_en  = out;                                   // [B,H,TY]
    float* logp  = out + (size_t)BB * HH * TY;            // [B,TX,TY]
    float* attn  = logp + (size_t)BB * TX * TY;           // [B,TX,TY]
    float* dr    = attn + (size_t)BB * TX * TY;           // [B,TX]

    char* ws = (char*)d_ws;
    _Float16* Apack  = (_Float16*)(ws + WS_A_OFF);
    _Float16* Bpack  = (_Float16*)(ws + WS_B_OFF);
    float*    constx = (float*)(ws + WS_CONST_OFF);
    int*      Xmap   = (int*)(ws + WS_XMAP_OFF);

    k_pack_a<<<(BB * TX + 255) / 256, 256, 0, stream>>>(mu, ls, Apack, constx);
    k_pack_b<<<(BB * TY + 255) / 256, 256, 0, stream>>>(yv, Bpack);

    // 4096 tiles total, 8 waves (tiles) per 256-thread block
    k_logp_wmma<<<(BB * (TX / 16) * (TY / 16)) / 8, 256, 0, stream>>>(
        Apack, Bpack, constx, logp);

    k_mas<<<BB, 256, 0, stream>>>(logp, xlen, ylen, attn, dr, Xmap);

    k_expand<<<(BB * HH * TY + 255) / 256, 256, 0, stream>>>(en, Xmap, o_en);
}
